// APG_MLP_15470472201006
// MI455X (gfx1250) — matrix operand order repaired and checked by reading, not compiled, not run on hardware
//
#include <hip/hip_runtime.h>
#include <hip/hip_bf16.h>

typedef __attribute__((ext_vector_type(16))) __bf16 v16bf;
typedef __attribute__((ext_vector_type(8)))  float  v8f;

__device__ __forceinline__ unsigned short f2bf(float f) {
    union { float f; unsigned u; } x; x.f = f;
    unsigned u = x.u;
    u += 0x7FFFu + ((u >> 16) & 1u);       // round-to-nearest-even
    return (unsigned short)(u >> 16);
}
__device__ __forceinline__ float bf2f(unsigned short h) {
    union { unsigned u; float f; } x; x.u = (unsigned)h << 16;
    return x.f;
}

union FragAB { v16bf v; unsigned u[8]; };

// 16B global -> LDS async copy (ASYNCcnt-tracked, no VGPR round trip).
__device__ __forceinline__ void async_cp_b128(unsigned lds_off, const unsigned short* g) {
    asm volatile("global_load_async_to_lds_b128 %0, %1, off"
                 :: "v"(lds_off), "v"(g) : "memory");
}
__device__ __forceinline__ void wait_async0() {
    asm volatile("s_wait_asynccnt 0x0" ::: "memory");
}
__device__ __forceinline__ unsigned lds_addr(const void* p) {
    return (unsigned)(unsigned long long)p;   // generic LDS addr[31:0] == LDS offset
}

// ---------------------------------------------------------------------------
// f32 -> bf16 conversion (plain, and transposing KxN -> NxK for weights)
// ---------------------------------------------------------------------------
__global__ __launch_bounds__(256)
void cvt_bf16(const float* __restrict__ in, unsigned short* __restrict__ out) {
    int i = blockIdx.x * 256 + threadIdx.x;
    out[i] = f2bf(in[i]);
}
__global__ __launch_bounds__(256)
void cvt_bf16_t(const float* __restrict__ in, unsigned short* __restrict__ out,
                int Kd, int N) {
    int i = blockIdx.x * 256 + threadIdx.x;   // i < Kd*N
    int k = i / N, n = i % N;
    out[(size_t)n * Kd + k] = f2bf(in[i]);
}

// ---------------------------------------------------------------------------
// bf16 GEMM via WMMA, tiles staged with global_load_async_to_lds_b128.
//   C[m,n] = sum_k A[m,k]*Wt[n,k] (+ vbias[n]) (+ mbias[m,n]) (relu) (f32|bf16 out)
// A: M x Kdim bf16 row-major.  Wt: N x Kdim bf16 row-major (pre-transposed).
// Block 256 thr = 8 waves; tile 64(M) x 64(N); wave = 32x16 (2 WMMA / K-step).
// LDS pitch 40 elems (80B): 16B async chunks aligned; 4B frag reads conflict-free.
// ---------------------------------------------------------------------------
__global__ __launch_bounds__(256)
void gemm_bf16_wmma(const unsigned short* __restrict__ A, int lda,
                    const unsigned short* __restrict__ Wt,
                    const float* __restrict__ vbias,
                    const float* __restrict__ mbias, int ldb,
                    void* __restrict__ Cout, int ldo,
                    int Kdim, int relu, int outBf16)
{
    __shared__ __align__(16) unsigned short sA[64][40];   // [row][k]
    __shared__ __align__(16) unsigned short sB[64][40];   // [col][k]

    const int tid    = threadIdx.x;
    const int m0     = blockIdx.x * 64;
    const int n0     = blockIdx.y * 64;
    const int waveId = tid >> 5, lane = tid & 31;
    const int wm     = waveId >> 2;        // 0..1 -> M offset wm*32
    const int wn     = waveId & 3;         // 0..3 -> N offset wn*16
    const int row    = lane & 15, khalf = lane >> 4;

    const int cr = tid >> 2, cc = tid & 3;              // copy row / 16B chunk
    const unsigned dA = lds_addr(&sA[0][0]) + cr * 80 + cc * 16;
    const unsigned dB = lds_addr(&sB[0][0]) + cr * 80 + cc * 16;

    v8f acc0 = {}; v8f acc1 = {};

    for (int kk = 0; kk < Kdim; kk += 32) {
        async_cp_b128(dA, A  + (size_t)(m0 + cr) * lda  + kk + cc * 8);
        async_cp_b128(dB, Wt + (size_t)(n0 + cr) * Kdim + kk + cc * 8);
        if (kk + 32 < Kdim)
            __builtin_prefetch(Wt + (size_t)(n0 + cr) * Kdim + kk + 32, 0, 1);
        wait_async0();
        __syncthreads();

        FragAB b;
        #pragma unroll
        for (int p = 0; p < 8; ++p)
            b.u[p] = *(const unsigned*)&sB[wn * 16 + row][khalf * 16 + 2 * p];

        FragAB a0, a1;
        #pragma unroll
        for (int p = 0; p < 8; ++p) {
            int kb = khalf * 16 + 2 * p;  // {0|16}+2*p
            a0.u[p] = *(const unsigned*)&sA[wm * 32 + row][kb];
            a1.u[p] = *(const unsigned*)&sA[wm * 32 + 16 + row][kb];
        }
        acc0 = __builtin_amdgcn_wmma_f32_16x16x32_bf16(false, a0.v, false, b.v,
                                                       (short)0, acc0, false, false);
        acc1 = __builtin_amdgcn_wmma_f32_16x16x32_bf16(false, a1.v, false, b.v,
                                                       (short)0, acc1, false, false);
        __syncthreads();
    }

    float* outF = (float*)Cout;
    unsigned short* outH = (unsigned short*)Cout;
    const int n  = n0 + wn * 16 + (lane & 15);
    const float vb = vbias ? vbias[n] : 0.f;
    #pragma unroll
    for (int r = 0; r < 8; ++r) {
        int mrow = (lane & 16) ? (r + 8) : r;
        int m    = m0 + wm * 32 + mrow;
        #pragma unroll
        for (int s = 0; s < 2; ++s) {
            int mm   = m + s * 16;
            float v  = (s ? acc1[r] : acc0[r]) + vb;
            if (mbias) v += mbias[(size_t)mm * ldb + n];
            if (relu)  v = fmaxf(v, 0.f);
            size_t idx = (size_t)mm * ldo + n;
            if (outBf16) outH[idx] = f2bf(v); else outF[idx] = v;
        }
    }
}

// ---------------------------------------------------------------------------
// Fused S-generation + contraction (S never touches HBM):
//   h2[b,j] = sum_k h[b,k] * ( h1[b,:] . hW2[:, 512+64k+j] + hb2[512+64k+j] )
// h1 tile async-staged to LDS; B fragments are contiguous b32 pairs from the
// transposed, L2-resident hW2t (4608 x 256 bf16). Per-wave jblk is constant;
// cross-wave k-partials reduced with LDS ds_add_f32.
// ---------------------------------------------------------------------------
__global__ __launch_bounds__(256)
void s_contract(const unsigned short* __restrict__ h1,   // B x 256 bf16
                const unsigned short* __restrict__ hW2t, // 4608 x 256 bf16 (N x K)
                const float* __restrict__ hb2,           // 4608 f32
                const float* __restrict__ hvec,          // B x 64 f32 (h = x@U)
                unsigned short* __restrict__ h2)         // B x 64 bf16
{
    __shared__ __align__(16) unsigned short sH1[16][264];  // 528B pitch (16B-mult)
    __shared__ float sH[16 * 64];
    __shared__ float sAcc[16 * 64];

    const int tid    = threadIdx.x;
    const int b0     = blockIdx.x * 16;
    const int waveId = tid >> 5, lane = tid & 31;
    const int row    = lane & 15, khalf = lane >> 4;

    const unsigned dH1 = lds_addr(&sH1[0][0]);
    #pragma unroll
    for (int q = 0; q < 2; ++q) {                      // 16x256 bf16 = 512 chunks
        int cid = tid + q * 256;
        int r = cid >> 5, ch = cid & 31;
        async_cp_b128(dH1 + r * 528 + ch * 16,
                      h1 + (size_t)(b0 + r) * 256 + ch * 8);
    }
    #pragma unroll
    for (int q = 0; q < 4; ++q) {
        int e = tid + q * 256;                         // 0..1023
        sH[e]   = hvec[(size_t)b0 * 64 + e];
        sAcc[e] = 0.f;
    }
    wait_async0();
    __syncthreads();

    v8f acc = {};
    for (int it = 0; it < 32; ++it) {
        int t  = waveId + it * 8;                      // 0..255 column chunks
        int n0 = t * 16;
        int k  = n0 >> 6;                              // fixed k within chunk
        v8f c = {};
        for (int kk = 0; kk < 256; kk += 32) {
            FragAB a, b;
            #pragma unroll
            for (int p = 0; p < 8; ++p) {
                int kb = khalf * 16 + 2 * p;
                a.u[p] = *(const unsigned*)&sH1[row][kk + kb];
                int kr = kk + khalf * 16 + 2 * p;      // even -> contiguous pair
                b.u[p] = *(const unsigned*)&hW2t[(size_t)(512 + n0 + row) * 256 + kr];
            }
            c = __builtin_amdgcn_wmma_f32_16x16x32_bf16(false, a.v, false, b.v,
                                                        (short)0, c, false, false);
        }
        float bcol = hb2[512 + n0 + row];
        #pragma unroll
        for (int r8 = 0; r8 < 8; ++r8) {
            int m = (lane & 16) ? (r8 + 8) : r8;
            acc[r8] += sH[m * 64 + k] * (c[r8] + bcol);
        }
    }
    int jbase = (waveId & 3) * 16;
    #pragma unroll
    for (int r8 = 0; r8 < 8; ++r8) {
        int m = (lane & 16) ? (r8 + 8) : r8;
        atomicAdd(&sAcc[m * 64 + jbase + row], acc[r8]);   // ds_add_f32
    }
    __syncthreads();
    #pragma unroll
    for (int q = 0; q < 4; ++q) {
        int e = tid + q * 256;
        h2[(size_t)b0 * 64 + e] = f2bf(sAcc[e]);
    }
}

// ---------------------------------------------------------------------------
// y[b] = dot(x[b, 0:512], Wout) + bout — one wave32 per row, shfl reduction.
// ---------------------------------------------------------------------------
__global__ __launch_bounds__(256)
void out_proj(const unsigned short* __restrict__ x, const float* __restrict__ Wout,
              const float* __restrict__ bout, float* __restrict__ y)
{
    int b    = blockIdx.x * 8 + (threadIdx.x >> 5);
    int lane = threadIdx.x & 31;
    float acc = 0.f;
    #pragma unroll
    for (int i = 0; i < 16; ++i)
        acc += bf2f(x[(size_t)b * 512 + lane + i * 32]) * Wout[lane + i * 32];
    #pragma unroll
    for (int off = 16; off > 0; off >>= 1)
        acc += __shfl_xor(acc, off, 32);
    if (lane == 0) y[b] = acc + bout[0];
}

extern "C" void kernel_launch(void* const* d_in, const int* in_sizes, int n_in,
                              void* d_out, int out_size, void* d_ws, size_t ws_size,
                              hipStream_t stream)
{
    (void)in_sizes; (void)n_in; (void)out_size; (void)ws_size;
    const int B = 8192, D = 512, K = 64, H = 256;
    const int S2 = K * K + D;   // 4608

    const float* x    = (const float*)d_in[0];
    const float* Wout = (const float*)d_in[19];
    const float* bout = (const float*)d_in[20];

    char* ws = (char*)d_ws;
    size_t off = 0;
    auto carve = [&](size_t bytes) {
        void* p = ws + off;
        off += (bytes + 255) & ~(size_t)255;
        return p;
    };
    unsigned short* xbf  = (unsigned short*)carve((size_t)B * D * 2);
    unsigned short* xb0  = (unsigned short*)carve((size_t)B * D * 2);
    unsigned short* xb1  = (unsigned short*)carve((size_t)B * D * 2);
    unsigned short* h1b  = (unsigned short*)carve((size_t)B * H * 2);
    float*          bias = (float*)         carve((size_t)B * D * 4);
    float*          hbuf = (float*)         carve((size_t)B * K * 4);
    unsigned short* h2b  = (unsigned short*)carve((size_t)B * K * 2);
    unsigned short* wT[3][4];   // per layer: hW1t(HxD), hW2t(S2xH), Ut(KxD), Vt(DxK)
    for (int l = 0; l < 3; ++l) {
        wT[l][0] = (unsigned short*)carve((size_t)D * H * 2);
        wT[l][1] = (unsigned short*)carve((size_t)H * S2 * 2);
        wT[l][2] = (unsigned short*)carve((size_t)D * K * 2);
        wT[l][3] = (unsigned short*)carve((size_t)K * D * 2);
    }

    dim3 blk(256);
    // one-time (per launch) conversions: x -> bf16; weights -> bf16 transposed NxK
    cvt_bf16<<<dim3(B * D / 256), blk, 0, stream>>>(x, xbf);
    for (int l = 0; l < 3; ++l) {
        const float* U   = (const float*)d_in[1 + l * 6 + 0];
        const float* V   = (const float*)d_in[1 + l * 6 + 1];
        const float* hW1 = (const float*)d_in[1 + l * 6 + 2];
        const float* hW2 = (const float*)d_in[1 + l * 6 + 4];
        cvt_bf16_t<<<dim3(D * H  / 256), blk, 0, stream>>>(hW1, wT[l][0], D, H);
        cvt_bf16_t<<<dim3(H * S2 / 256), blk, 0, stream>>>(hW2, wT[l][1], H, S2);
        cvt_bf16_t<<<dim3(D * K  / 256), blk, 0, stream>>>(U,   wT[l][2], D, K);
        cvt_bf16_t<<<dim3(K * D  / 256), blk, 0, stream>>>(V,   wT[l][3], K, D);
    }

    const unsigned short* cur = xbf;
    unsigned short* nxt = xb0;
    for (int l = 0; l < 3; ++l) {
        const float* hb1 = (const float*)d_in[1 + l * 6 + 3];
        const float* hb2 = (const float*)d_in[1 + l * 6 + 5];

        // h1 = relu(x @ hW1 + hb1)              (8192x512)x(512x256) -> bf16
        gemm_bf16_wmma<<<dim3(B / 64, H / 64), blk, 0, stream>>>(
            cur, D, wT[l][0], hb1, nullptr, 0, h1b, H, D, 1, 1);
        // bias = h1 @ hW2[:,0:512] + hb2[0:512] (8192x256)x(256x512) -> f32
        gemm_bf16_wmma<<<dim3(B / 64, D / 64), blk, 0, stream>>>(
            h1b, H, wT[l][1], hb2, nullptr, 0, bias, D, H, 0, 0);
        // h = x @ U                             (8192x512)x(512x64) -> f32
        gemm_bf16_wmma<<<dim3(B / 64, K / 64), blk, 0, stream>>>(
            cur, D, wT[l][2], nullptr, nullptr, 0, hbuf, K, D, 0, 0);
        // h2 = einsum(h, S), S generated in-register
        s_contract<<<dim3(B / 16), blk, 0, stream>>>(h1b, wT[l][1], hb2, hbuf, h2b);
        // x' = relu(h2 @ V + bias)              (8192x64)x(64x512) -> bf16
        gemm_bf16_wmma<<<dim3(B / 64, D / 64), blk, 0, stream>>>(
            h2b, K, wT[l][3], nullptr, bias, D, nxt, D, K, 1, 1);

        cur = nxt;
        nxt = (nxt == xb0) ? xb1 : xb0;
    }
    out_proj<<<dim3(B / 8), blk, 0, stream>>>(cur, Wout, bout, (float*)d_out);
}
